// Llama4TextExpertsLinear_25477746000406
// MI455X (gfx1250) — matrix operand order repaired and checked by reading, not compiled, not run on hardware
//
#include <hip/hip_runtime.h>

#define E_ 8
#define H_ 2048
#define I_ 8192
#define I2_ 16384
#define T_ 1024

typedef unsigned int uint32;
typedef unsigned short ushort16;
typedef __attribute__((ext_vector_type(8)))  __bf16 v8bf;
typedef __attribute__((ext_vector_type(16))) __bf16 v16bf;
typedef __attribute__((ext_vector_type(8)))  float  v8f;

// ---- helpers -------------------------------------------------------------

__device__ __forceinline__ ushort16 f2bf(float f) {
    uint32 u = __builtin_bit_cast(uint32, f);
    u += 0x7FFFu + ((u >> 16) & 1u);       // round-to-nearest-even
    return (ushort16)(u >> 16);
}
__device__ __forceinline__ uint32 pk2bf(float a, float b) {
    return (uint32)f2bf(a) | ((uint32)f2bf(b) << 16);
}

// A-fragment (16x32 bf16) contiguous k: lane<16 -> K[0..15]; lane>=16 ->
// K[16..31].  p points at (row, c0) with c0 = 0 or 16 elements.
__device__ __forceinline__ v16bf ld_fragA(const ushort16* p) {
    v8bf lo = *(const v8bf*)p;           // K c0..c0+7
    v8bf hi = *(const v8bf*)(p + 8);     // K c0+8..c0+15
    return __builtin_shufflevector(lo, hi, 0,1,2,3,4,5,6,7,8,9,10,11,12,13,14,15);
}
// B-fragment (32x16 bf16) in contiguous k order: lanes 0-15 hold K=0..15, lanes 16-31 hold
// K=16..31, packed pairs per VGPR -> 32 contiguous bytes per lane.
__device__ __forceinline__ v16bf ld_fragB(const ushort16* p) {
    v8bf lo = *(const v8bf*)p;           // K c0..c0+7
    v8bf hi = *(const v8bf*)(p + 8);     // K c0+8..c0+15
    return __builtin_shufflevector(lo, hi, 0,1,2,3,4,5,6,7,8,9,10,11,12,13,14,15);
}

// ---- kernel 0: fp32 -> bf16 convert of hidden_states ---------------------

__global__ __launch_bounds__(256) void cvt_bf16(const float* __restrict__ in,
                                                ushort16* __restrict__ out, int n) {
    int i = (blockIdx.x * 256 + threadIdx.x) * 4;
    if (i < n) {
        float4 v = *(const float4*)(in + i);
        uint2 pp; pp.x = pk2bf(v.x, v.y); pp.y = pk2bf(v.z, v.w);
        *(uint2*)(out + i) = pp;
    }
}

// ---- kernel 1: gate_up GEMM + fused SiLU, bf16 h output ------------------
// grid: (I_/128, T_/128, E_), block 256 threads = 8 waves.
// waves 0..3 -> gate 128x128 tile, waves 4..7 -> up 128x128 tile.
// Double-buffered LDS tiles (2 x 30720 B), epilogue reuses the same union.

__global__ __launch_bounds__(256) void gemm1_silu(const ushort16* __restrict__ Xb,
                                                  const float* __restrict__ W1,
                                                  ushort16* __restrict__ Hb) {
    const int e  = blockIdx.z;
    const int t0 = blockIdx.y * 128;
    const int n0 = blockIdx.x * 128;
    const float* W = W1 + (size_t)e * H_ * I2_;

    __shared__ __align__(16) char smem[67584];    // max(2x tiles 61440B, epilogue 67584B)
    float* sEp = (float*)smem;                    // epilogue 128 x 132 fp32

    const int tid  = threadIdx.x;
    const int lane = tid & 31;
    const int w    = tid >> 5;
    const int half = w >> 2;                      // 0=gate, 1=up
    const int wl   = w & 3;
    const int wm   = wl >> 1, wn = wl & 1;        // 2x2 waves over 128x128

    v8f acc[4][4];
#pragma unroll
    for (int i = 0; i < 4; ++i)
#pragma unroll
        for (int j = 0; j < 4; ++j)
            acc[i][j] = (v8f){0.f,0.f,0.f,0.f,0.f,0.f,0.f,0.f};

    // A-copy map: 512 16B-chunks, 2 per thread (same row)
    const int arow = tid >> 1;                    // 0..127
    const int ac   = (tid & 1) * 2;               // chunk 0 or 2
    // B-copy map: n = tid&127, k-half = tid>>7
    const int bn  = tid & 127;
    const int bkh = tid >> 7;

    const int c0a = (lane < 16) ? 0 : 16;
    const int c0b = (lane < 16) ? 0 : 16;
    const int mr  = lane & 15;

    auto load_tiles = [&](int k0, int p) {
        ushort16* dA  = (ushort16*)smem + p * 15360;    // 128 x 40
        ushort16* dBg = dA  + 5120;                     // 128 x 40 [n][k]
        ushort16* dBu = dBg + 5120;
        const uint4* ga = (const uint4*)(Xb + (size_t)(t0 + arow) * H_ + k0 + ac * 8);
        uint4 a0 = ga[0], a1 = ga[1];
        *(uint4*)(dA + arow * 40 + ac * 8)       = a0;
        *(uint4*)(dA + arow * 40 + (ac + 1) * 8) = a1;
        const float* wg = W + (size_t)(k0 + bkh * 16) * I2_ + n0 + bn;
        const float* wu = wg + I_;
#pragma unroll
        for (int kk = 0; kk < 8; ++kk) {
            *(uint32*)(dBg + bn * 40 + bkh * 16 + 2 * kk) =
                pk2bf(wg[(size_t)(2 * kk) * I2_], wg[(size_t)(2 * kk + 1) * I2_]);
            *(uint32*)(dBu + bn * 40 + bkh * 16 + 2 * kk) =
                pk2bf(wu[(size_t)(2 * kk) * I2_], wu[(size_t)(2 * kk + 1) * I2_]);
        }
    };

    const int nk = H_ / 32;
    load_tiles(0, 0);
    __syncthreads();

    for (int kt = 0; kt < nk; ++kt) {
        const int p = kt & 1;
        if (kt + 1 < nk) load_tiles((kt + 1) * 32, p ^ 1);   // overlaps with WMMAs below

        const ushort16* sA = (const ushort16*)smem + p * 15360;
        const ushort16* sB = sA + 5120 + half * 5120;        // gate or up tile
        v16bf af[4], bf[4];
#pragma unroll
        for (int i = 0; i < 4; ++i) {
            af[i] = ld_fragA(sA + (wm * 64 + i * 16 + mr) * 40 + c0a);
            bf[i] = ld_fragB(sB + (wn * 64 + i * 16 + mr) * 40 + c0b);
        }
#pragma unroll
        for (int i = 0; i < 4; ++i)
#pragma unroll
            for (int j = 0; j < 4; ++j)
                acc[i][j] = __builtin_amdgcn_wmma_f32_16x16x32_bf16(
                    false, af[i], false, bf[j], (short)0, acc[i][j], false, false);
        __syncthreads();
    }

    // Epilogue: up waves -> LDS, gate waves compute up*silu(gate) -> bf16 Hb
    const int rbase = (lane >= 16) ? 8 : 0;
    if (half == 1) {
#pragma unroll
        for (int i = 0; i < 4; ++i)
#pragma unroll
            for (int j = 0; j < 4; ++j)
#pragma unroll
                for (int v = 0; v < 8; ++v) {
                    int row = wm * 64 + i * 16 + v + rbase;
                    int col = wn * 64 + j * 16 + mr;
                    sEp[row * 132 + col] = acc[i][j][v];
                }
    }
    __syncthreads();
    if (half == 0) {
        ushort16* hb = Hb + (size_t)e * T_ * I_ + (size_t)t0 * I_ + n0;
#pragma unroll
        for (int i = 0; i < 4; ++i)
#pragma unroll
            for (int j = 0; j < 4; ++j)
#pragma unroll
                for (int v = 0; v < 8; ++v) {
                    int row = wm * 64 + i * 16 + v + rbase;
                    int col = wn * 64 + j * 16 + mr;
                    float g = acc[i][j][v];
                    float u = sEp[row * 132 + col];
                    float s = g / (1.f + __expf(-g));
                    hb[(size_t)row * I_ + col] = f2bf(u * s);
                }
    }
}

// ---- kernel 2: down GEMM, fp32 out ---------------------------------------
// grid: (H_/256, T_/128, E_), block 256 = 8 waves, 128x256 tile, K=8192.
// Double-buffered LDS (2 x 30720 B).

__global__ __launch_bounds__(256) void gemm2_down(const ushort16* __restrict__ Hb,
                                                  const float* __restrict__ W2,
                                                  float* __restrict__ Out) {
    const int e  = blockIdx.z;
    const int t0 = blockIdx.y * 128;
    const int n0 = blockIdx.x * 256;
    const ushort16* A = Hb + (size_t)e * T_ * I_;
    const float*    W = W2 + (size_t)e * I_ * H_;

    __shared__ __align__(16) ushort16 smem2[2 * 15360];     // per buffer: A 128x40, B 256x40

    const int tid  = threadIdx.x;
    const int lane = tid & 31;
    const int w    = tid >> 5;
    const int wm   = w >> 2, wn = w & 3;          // 2x4 waves over 128x256

    v8f acc[4][4];
#pragma unroll
    for (int i = 0; i < 4; ++i)
#pragma unroll
        for (int j = 0; j < 4; ++j)
            acc[i][j] = (v8f){0.f,0.f,0.f,0.f,0.f,0.f,0.f,0.f};

    const int arow = tid >> 1;
    const int ac   = (tid & 1) * 2;
    const int bn   = tid;                         // 0..255
    const int c0a  = (lane < 16) ? 0 : 16;
    const int c0b  = (lane < 16) ? 0 : 16;
    const int mr   = lane & 15;

    auto load_tiles = [&](int k0, int p) {
        ushort16* dA = smem2 + p * 15360;
        ushort16* dB = dA + 5120;
        const uint4* ga = (const uint4*)(A + (size_t)(t0 + arow) * I_ + k0 + ac * 8);
        uint4 a0 = ga[0], a1 = ga[1];
        *(uint4*)(dA + arow * 40 + ac * 8)       = a0;
        *(uint4*)(dA + arow * 40 + (ac + 1) * 8) = a1;
        const float* wp = W + (size_t)k0 * H_ + n0 + bn;
#pragma unroll
        for (int kk = 0; kk < 16; ++kk) {
            *(uint32*)(dB + bn * 40 + 2 * kk) =
                pk2bf(wp[(size_t)(2 * kk) * H_], wp[(size_t)(2 * kk + 1) * H_]);
        }
    };

    const int nk = I_ / 32;
    load_tiles(0, 0);
    __syncthreads();

    for (int kt = 0; kt < nk; ++kt) {
        const int p = kt & 1;
        if (kt + 1 < nk) load_tiles((kt + 1) * 32, p ^ 1);

        const ushort16* sA = smem2 + p * 15360;
        const ushort16* sB = sA + 5120;
        v16bf af[4], bf[4];
#pragma unroll
        for (int i = 0; i < 4; ++i) {
            af[i] = ld_fragA(sA + (wm * 64 + i * 16 + mr) * 40 + c0a);
            bf[i] = ld_fragB(sB + (wn * 64 + i * 16 + mr) * 40 + c0b);
        }
#pragma unroll
        for (int i = 0; i < 4; ++i)
#pragma unroll
            for (int j = 0; j < 4; ++j)
                acc[i][j] = __builtin_amdgcn_wmma_f32_16x16x32_bf16(
                    false, af[i], false, bf[j], (short)0, acc[i][j], false, false);
        __syncthreads();
    }

    float* ob = Out + ((size_t)e * T_ + t0) * H_ + n0;
    const int rbase = (lane >= 16) ? 8 : 0;
#pragma unroll
    for (int i = 0; i < 4; ++i)
#pragma unroll
        for (int j = 0; j < 4; ++j)
#pragma unroll
            for (int v = 0; v < 8; ++v) {
                int row = wm * 64 + i * 16 + v + rbase;
                int col = wn * 64 + j * 16 + mr;
                ob[(size_t)row * H_ + col] = acc[i][j][v];
            }
}

// ---- launch ---------------------------------------------------------------

extern "C" void kernel_launch(void* const* d_in, const int* in_sizes, int n_in,
                              void* d_out, int out_size, void* d_ws, size_t ws_size,
                              hipStream_t stream) {
    const float* hs = (const float*)d_in[0];          // (T,H) fp32
    const float* w1 = (const float*)d_in[1];          // (E,H,2I) fp32
    const float* w2 = (const float*)d_in[2];          // (E,I,H) fp32

    ushort16* Xb = (ushort16*)d_ws;                                       // T*H bf16 (4 MB)
    ushort16* Hb = (ushort16*)((char*)d_ws + (size_t)T_ * H_ * 2);        // E*T*I bf16 (128 MB)

    cvt_bf16<<<(T_ * H_) / (256 * 4), 256, 0, stream>>>(hs, Xb, T_ * H_);

    dim3 g1(I_ / 128, T_ / 128, E_);
    gemm1_silu<<<g1, 256, 0, stream>>>(Xb, w1, Hb);

    dim3 g2(H_ / 256, T_ / 128, E_);
    gemm2_down<<<g2, 256, 0, stream>>>(Hb, w2, (float*)d_out);
}
